// Affine_83811991814659
// MI455X (gfx1250) — matrix operand order repaired and checked by reading, not compiled, not run on hardware
//
#include <hip/hip_runtime.h>

// ---------------------------------------------------------------------------
// MoE hard-routed linear for MI455X (gfx1250, wave32, WMMA).
//   out[t] = x[t] @ W[p[t]]^T + b[p[t]]
// Pipeline: histogram -> scan -> scatter (bucket tokens by expert) ->
//           grouped GEMM with split-f16 (hi+lo) WMMA, f32 accumulate.
// This revision software-pipelines the global->LDS staging (register
// double-buffer) so HBM latency overlaps the WMMA compute phase.
// ---------------------------------------------------------------------------

typedef __attribute__((ext_vector_type(16))) _Float16 v16h;
typedef __attribute__((ext_vector_type(8)))  _Float16 v8h;
typedef __attribute__((ext_vector_type(8)))  float    v8f;

#define T_TOK   32768
#define DIM_IN  512
#define DIM_OUT 512
#define NEXP    8
#define TM      128          // rows (tokens) per block tile
#define TN      128          // cols (outputs) per block tile
#define KC      32           // K per LDS chunk == one WMMA K step
#define LDH     40           // padded LDS K-stride in halves (32+8): conflict-free, 16B aligned

// ---------------- routing kernels ----------------

__global__ void zero_counts_kernel(int* counts) {
  if (threadIdx.x < NEXP) counts[threadIdx.x] = 0;
}

__global__ void hist_kernel(const int* __restrict__ parts, int* __restrict__ counts) {
  int t = blockIdx.x * blockDim.x + threadIdx.x;
  if (t < T_TOK) atomicAdd(&counts[parts[t]], 1);
}

__global__ void scan_kernel(const int* __restrict__ counts, int* __restrict__ offs,
                            int* __restrict__ cursors) {
  if (threadIdx.x == 0) {
    int acc = 0;
    for (int i = 0; i < NEXP; ++i) { offs[i] = acc; cursors[i] = acc; acc += counts[i]; }
    offs[NEXP] = acc;
  }
}

__global__ void scatter_kernel(const int* __restrict__ parts, int* __restrict__ cursors,
                               int* __restrict__ idxbuf) {
  int t = blockIdx.x * blockDim.x + threadIdx.x;
  if (t < T_TOK) {
    int pos = atomicAdd(&cursors[parts[t]], 1);
    idxbuf[pos] = t;
  }
}

// ---------------- grouped GEMM ----------------

// Load a 16-half WMMA fragment from LDS: two aligned 16-byte (b128) loads.
__device__ __forceinline__ v16h frag16(const _Float16* p, int o0, int o1) {
  v8h a = *(const v8h*)(p + o0);
  v8h b = *(const v8h*)(p + o1);
  v16h r;
#pragma unroll
  for (int i = 0; i < 8; ++i) { r[i] = a[i]; r[i + 8] = b[i]; }
  return r;
}

// Split one float4 into hi/lo f16 quads and store to LDS.
__device__ __forceinline__ void split_store(const float4 v, _Float16* ph, _Float16* pl) {
  _Float16 h0 = (_Float16)v.x, h1 = (_Float16)v.y;
  _Float16 h2 = (_Float16)v.z, h3 = (_Float16)v.w;
  ph[0] = h0; ph[1] = h1; ph[2] = h2; ph[3] = h3;
  pl[0] = (_Float16)(v.x - (float)h0);
  pl[1] = (_Float16)(v.y - (float)h1);
  pl[2] = (_Float16)(v.z - (float)h2);
  pl[3] = (_Float16)(v.w - (float)h3);
}

__global__ __launch_bounds__(256)
void moe_gemm_kernel(const float* __restrict__ x, const int* __restrict__ idxbuf,
                     const int* __restrict__ offs, const float* __restrict__ W,
                     const float* __restrict__ bias, float* __restrict__ out) {
  __shared__ __align__(16) _Float16 sAh[TM * LDH];
  __shared__ __align__(16) _Float16 sAl[TM * LDH];
  __shared__ __align__(16) _Float16 sBh[TN * LDH];
  __shared__ __align__(16) _Float16 sBl[TN * LDH];   // 4 * 10 KB = 40 KB LDS

  const int e    = blockIdx.z;
  const int base = offs[e];
  const int cnt  = offs[e + 1] - base;
  const int mtiles = (cnt + TM - 1) / TM;
  if (blockIdx.x >= mtiles) return;                 // uniform per block: EXEC stays all-1s
  const int tm = blockIdx.x;
  const int tn = blockIdx.y;

  const int tid  = threadIdx.x;
  const int lane = tid & 31;
  const int ln   = lane & 15;       // N (for B/C/D) or M (for A) within subtile
  const int hf   = lane >> 4;       // lane half selects K/M groups
  const int wid  = tid >> 5;        // 8 waves
  const int wm   = wid & 3;         // 4 waves over M: 32 rows each
  const int wn   = wid >> 2;        // 2 waves over N: 64 cols each

  v8f acc[2][4] = {};               // 2 M-subtiles x 4 N-subtiles per wave

  const float* Wp = W + ((size_t)e * DIM_OUT + (size_t)tn * TN) * DIM_IN;

  // ---- loop-invariant staging coordinates: 4 float4 slots per thread ----
  const float* aPtr[4];             // gathered token row base (+k4)
  const float* bPtr[4];             // W row base (+k4)
  int          ldsOff[4];           // LDS destination offset (halves)
#pragma unroll
  for (int i = 0; i < 4; ++i) {
    int it  = tid + i * 256;
    int row = it >> 3;              // 0..127
    int k4  = (it & 7) * 4;         // 0,4,...,28
    int rg  = tm * TM + row;
    int rc  = rg < cnt ? rg : cnt - 1;
    int tok = idxbuf[base + rc];
    aPtr[i]   = x + (size_t)tok * DIM_IN + k4;
    bPtr[i]   = Wp + (size_t)row * DIM_IN + k4;
    ldsOff[i] = row * LDH + k4;
  }

  // ---- preload chunk 0 into registers ----
  float4 ra[4], rb[4];
#pragma unroll
  for (int i = 0; i < 4; ++i) {
    ra[i] = *(const float4*)(aPtr[i]);
    rb[i] = *(const float4*)(bPtr[i]);
  }

  for (int kc = 0; kc < DIM_IN; kc += KC) {
    // ---- convert current registers -> split-f16 LDS tiles ----
#pragma unroll
    for (int i = 0; i < 4; ++i) {
      split_store(ra[i], sAh + ldsOff[i], sAl + ldsOff[i]);
      split_store(rb[i], sBh + ldsOff[i], sBl + ldsOff[i]);
    }
    __syncthreads();

    // ---- issue next chunk's global loads (overlap with frag loads + WMMA) ----
    if (kc + KC < DIM_IN) {
#pragma unroll
      for (int i = 0; i < 4; ++i) {
        ra[i] = *(const float4*)(aPtr[i] + kc + KC);
        rb[i] = *(const float4*)(bPtr[i] + kc + KC);
      }
    }

    // ---- fragments: documented 16-bit A 16x32 layout; B 32x16 layout ----
    v16h ah[2], al[2], bh[4], bl[4];
#pragma unroll
    for (int s = 0; s < 2; ++s) {
      int r  = wm * 32 + s * 16 + ln;        // A: lane = M row
      int o0 = hf * 8;                        // VGPR0-3: K = 8*half + 0..7
      int o1 = 16 + hf * 8;                   // VGPR4-7: K = 16 + 8*half + 0..7
      ah[s] = frag16(sAh + r * LDH, o0, o1);
      al[s] = frag16(sAl + r * LDH, o0, o1);
    }
#pragma unroll
    for (int j = 0; j < 4; ++j) {
      int c  = wn * 64 + j * 16 + ln;        // B: lane = N col
      int o0 = hf * 8;                        // VGPR0-3: K = 8*half + 0..7
      int o1 = 16 + hf * 8;                   // VGPR4-7: K = 16 + 8*half + 0..7
      bh[j] = frag16(sBh + c * LDH, o0, o1);
      bl[j] = frag16(sBl + c * LDH, o0, o1);
    }
    __syncthreads();

    // ---- split-f16 triple product: x*W ~= xh*Wh + xh*Wl + xl*Wh ----
#pragma unroll
    for (int s = 0; s < 2; ++s)
#pragma unroll
      for (int j = 0; j < 4; ++j) {
        acc[s][j] = __builtin_amdgcn_wmma_f32_16x16x32_f16(
            false, ah[s], false, bh[j], (short)0, acc[s][j], false, false);
        acc[s][j] = __builtin_amdgcn_wmma_f32_16x16x32_f16(
            false, ah[s], false, bl[j], (short)0, acc[s][j], false, false);
        acc[s][j] = __builtin_amdgcn_wmma_f32_16x16x32_f16(
            false, al[s], false, bh[j], (short)0, acc[s][j], false, false);
      }
  }

  // ---- epilogue: bias + scatter back to token rows ----
  // Cache token ids / validity per output row (reused by all 4 j-subtiles).
  int  toks[2][8];
  bool valid[2][8];
#pragma unroll
  for (int s = 0; s < 2; ++s)
#pragma unroll
    for (int r = 0; r < 8; ++r) {
      int rowg   = tm * TM + wm * 32 + s * 16 + hf * 8 + r;   // C/D: VGPR r -> M = r + 8*half
      valid[s][r] = rowg < cnt;
      int rc      = rowg < cnt ? rowg : cnt - 1;
      toks[s][r]  = idxbuf[base + rc];
    }

  const int colbase = tn * TN + wn * 64;
#pragma unroll
  for (int s = 0; s < 2; ++s) {
#pragma unroll
    for (int j = 0; j < 4; ++j) {
      int col  = colbase + j * 16 + ln;
      float bv = bias[e * DIM_OUT + col];
#pragma unroll
      for (int r = 0; r < 8; ++r) {
        if (valid[s][r]) {
          out[(size_t)toks[s][r] * DIM_OUT + col] = acc[s][j][r] + bv;
        }
      }
    }
  }
}

// ---------------- launch ----------------

extern "C" void kernel_launch(void* const* d_in, const int* in_sizes, int n_in,
                              void* d_out, int out_size, void* d_ws, size_t ws_size,
                              hipStream_t stream) {
  const float* x     = (const float*)d_in[0];
  const int*   parts = (const int*)d_in[1];
  const float* W     = (const float*)d_in[2];
  const float* b     = (const float*)d_in[3];
  float*       out   = (float*)d_out;

  int* counts  = (int*)d_ws;          // [8]
  int* offs    = counts + 8;          // [9]
  int* cursors = counts + 20;         // [8]
  int* idxbuf  = counts + 32;         // [32768]

  zero_counts_kernel<<<1, 32, 0, stream>>>(counts);
  hist_kernel<<<T_TOK / 256, 256, 0, stream>>>(parts, counts);
  scan_kernel<<<1, 32, 0, stream>>>(counts, offs, cursors);
  scatter_kernel<<<T_TOK / 256, 256, 0, stream>>>(parts, cursors, idxbuf);

  dim3 grid(T_TOK / TM, DIM_OUT / TN, NEXP);   // worst-case M tiles; blocks early-exit
  moe_gemm_kernel<<<grid, 256, 0, stream>>>(x, idxbuf, offs, W, b, out);
}
